// GATv2Backbone_51376398795483
// MI455X (gfx1250) — compile-verified, matrix operand order checked by reading, not run on hardware
//
#include <hip/hip_runtime.h>
#include <hip/hip_bf16.h>
#include <math.h>

// ---------------------------------------------------------------------------
// GATv2 backbone for MI455X (gfx1250, wave32).
// GEMMs use v_wmma_f32_16x16x32_bf16; edge softmax/scatter are memory-bound
// wave-per-edge kernels using f32 atomics (resolve in the 192MB L2).
// ---------------------------------------------------------------------------

typedef __attribute__((ext_vector_type(16))) __bf16 v16bf;
typedef __attribute__((ext_vector_type(8)))  float  v8f;

union BF16x16 { v16bf v; unsigned int u[8]; };

__device__ __forceinline__ unsigned short f2bf(float f) {
    unsigned int x = __float_as_uint(f);
    unsigned int r = x + 0x7FFFu + ((x >> 16) & 1u);   // round-to-nearest-even
    return (unsigned short)(r >> 16);
}

__device__ __forceinline__ float atomicMaxF(float* addr, float val) {
    int* ia = (int*)addr;
    int old = __float_as_int(*addr);
    while (__int_as_float(old) < val) {
        int assumed = old;
        old = atomicCAS(ia, assumed, __float_as_int(val));
        if (old == assumed) break;
    }
    return __int_as_float(old);
}

// -------------------- weight convert: W[k][n] f32 -> Wt[n][k] bf16 ----------
__global__ void conv_w_bf16t(const float* __restrict__ W, unsigned short* __restrict__ Wt) {
    int i = blockIdx.x * blockDim.x + threadIdx.x;
    if (i >= 128 * 128) return;
    int k = i >> 7, n = i & 127;
    Wt[n * 128 + k] = f2bf(W[i]);
}

// -------------------- generic fill ------------------------------------------
__global__ void fill_f32(float* __restrict__ p, float v, int n) {
    int i = blockIdx.x * blockDim.x + threadIdx.x;
    if (i < n) p[i] = v;
}

// -------------------- GEMM: D[M x 128] = A[M x 128] * W[128 x 128] + bias ---
// A f32 row-major; Wt bf16 pre-transposed [n][k]; block = 256 thr = 8 waves,
// each block owns a 128-row tile, each wave a 16x128 strip.
__global__ __launch_bounds__(256)
void gemm128_wmma_bf16(const float* __restrict__ A,
                       const unsigned short* __restrict__ Wt,
                       const float* __restrict__ bias,
                       float* __restrict__ D, int M) {
    __shared__ unsigned short sA[128 * 128];  // [row][k] bf16, 32KB
    __shared__ unsigned short sW[128 * 128];  // [n][k]  bf16, 32KB

    const int tid  = threadIdx.x;
    const int lane = tid & 31;
    const int wave = tid >> 5;
    const int rowBase = blockIdx.x * 128;

    // stage W (already bf16) as uint copies
    {
        const unsigned int* Wu = (const unsigned int*)Wt;
        unsigned int* sWu = (unsigned int*)sW;
        for (int i = tid; i < 128 * 128 / 2; i += 256) sWu[i] = Wu[i];
    }
    // stage A tile, f32 -> bf16 (float4 = global_load_b128)
    for (int i = tid; i < 128 * 128 / 4; i += 256) {
        int r  = i >> 5;            // 32 float4 per 128-wide row
        int c4 = (i & 31) * 4;
        int grow = rowBase + r;
        float4 f;
        if (grow < M) f = *(const float4*)(A + (size_t)grow * 128 + c4);
        else          { f.x = 0.f; f.y = 0.f; f.z = 0.f; f.w = 0.f; }
        sA[r * 128 + c4 + 0] = f2bf(f.x);
        sA[r * 128 + c4 + 1] = f2bf(f.y);
        sA[r * 128 + c4 + 2] = f2bf(f.z);
        sA[r * 128 + c4 + 3] = f2bf(f.w);
    }
    __syncthreads();

    v8f acc[8];
    #pragma unroll
    for (int t = 0; t < 8; ++t)
        #pragma unroll
        for (int e = 0; e < 8; ++e) acc[t][e] = 0.f;

    // ISA 16-bit A 16x32 layout: lanes 0-15 -> K pairs {0..3, 8..11},
    // lanes 16-31 -> K pairs {4..7, 12..15}  (pair = K/2)
    const int mrow  = (wave << 4) + (lane & 15);
    const int pbase = (lane < 16) ? 0 : 4;
    const int ncol0 = lane & 15;

    for (int kk = 0; kk < 128; kk += 32) {
        BF16x16 afrag;
        const unsigned int* rowp = (const unsigned int*)(sA + mrow * 128 + kk);
        #pragma unroll
        for (int v = 0; v < 4; ++v) {
            afrag.u[v]     = rowp[pbase + v];
            afrag.u[4 + v] = rowp[pbase + 8 + v];
        }
        #pragma unroll
        for (int t = 0; t < 8; ++t) {
            BF16x16 bfrag;
            const unsigned int* np =
                (const unsigned int*)(sW + (t * 16 + ncol0) * 128 + kk);
            #pragma unroll
            for (int v = 0; v < 4; ++v) {
                bfrag.u[v]     = np[pbase + v];
                bfrag.u[4 + v] = np[pbase + 8 + v];
            }
            acc[t] = __builtin_amdgcn_wmma_f32_16x16x32_bf16(
                false, afrag.v, false, bfrag.v, (short)0, acc[t], false, false);
        }
    }

    // C/D layout: VGPR r -> lanes 0-15: M=r, lanes 16-31: M=8+r; N = lane&15
    const int rowOff = (wave << 4) + ((lane < 16) ? 0 : 8);
    #pragma unroll
    for (int t = 0; t < 8; ++t) {
        int col = t * 16 + (lane & 15);
        float bc = bias[col];
        #pragma unroll
        for (int e = 0; e < 8; ++e) {
            int grow = rowBase + rowOff + e;
            if (grow < M) D[(size_t)grow * 128 + col] = acc[t][e] + bc;
        }
    }
}

// -------------------- pass B: per-edge logits + segment max -----------------
// one wave per edge; lane handles features [4*lane, 4*lane+4)
template <int H>
__global__ __launch_bounds__(256)
void edge_attn_max(const float* __restrict__ xl, const float* __restrict__ xr,
                   const int* __restrict__ src_arr, const int* __restrict__ dst_arr,
                   const float* __restrict__ att, float* __restrict__ logits,
                   float* __restrict__ mmax, int nE, int nN) {
    int wid  = (blockIdx.x * blockDim.x + threadIdx.x) >> 5;
    int lane = threadIdx.x & 31;
    if (wid >= nE + nN) return;
    int s, d;
    if (wid < nE) { s = src_arr[wid]; d = dst_arr[wid]; }
    else          { s = wid - nE;     d = s; }
    int f0 = lane * 4;
    float4 lv = *(const float4*)(xl + (size_t)s * 128 + f0);
    float4 rv = *(const float4*)(xr + (size_t)d * 128 + f0);
    float vals[4] = {lv.x + rv.x, lv.y + rv.y, lv.z + rv.z, lv.w + rv.w};
    float p = 0.f;
    #pragma unroll
    for (int j = 0; j < 4; ++j) {
        float v = vals[j];
        v = (v > 0.f) ? v : 0.2f * v;          // LeakyReLU
        p += v * att[f0 + j];
    }
    p += __shfl_xor(p, 1, 32);
    p += __shfl_xor(p, 2, 32);
    p += __shfl_xor(p, 4, 32);
    if (H == 4) {
        if ((lane & 7) == 0) {
            int hh = lane >> 3;
            logits[(size_t)wid * 4 + hh] = p;
            atomicMaxF(&mmax[(size_t)d * 4 + hh], p);
        }
    } else {
        p += __shfl_xor(p, 8, 32);
        p += __shfl_xor(p, 16, 32);
        if (lane == 0) {
            logits[wid] = p;
            atomicMaxF(&mmax[d], p);
        }
    }
}

// -------------------- pass C: a = exp(logit - max), denom += a --------------
template <int H>
__global__ void edge_softmax_num(const int* __restrict__ dst_arr,
                                 float* __restrict__ logits,
                                 const float* __restrict__ mmax,
                                 float* __restrict__ denom, int nE, int nN) {
    int i = blockIdx.x * blockDim.x + threadIdx.x;
    int total = (nE + nN) * H;
    if (i >= total) return;
    int e  = i / H;
    int hh = i - e * H;
    int d  = (e < nE) ? dst_arr[e] : e - nE;
    float a = __expf(logits[i] - mmax[(size_t)d * H + hh]);
    logits[i] = a;
    atomicAdd(&denom[(size_t)d * H + hh], a);
}

// -------------------- pass D: acc[dst] += xl[src] * a / denom[dst] ----------
template <int H>
__global__ __launch_bounds__(256)
void edge_scatter(const float* __restrict__ xl,
                  const int* __restrict__ src_arr, const int* __restrict__ dst_arr,
                  const float* __restrict__ logits, const float* __restrict__ denom,
                  float* __restrict__ acc, int nE, int nN) {
    int wid  = (blockIdx.x * blockDim.x + threadIdx.x) >> 5;
    int lane = threadIdx.x & 31;
    if (wid >= nE + nN) return;
    int s, d;
    if (wid < nE) { s = src_arr[wid]; d = dst_arr[wid]; }
    else          { s = wid - nE;     d = s; }
    int hh = (H == 4) ? (lane >> 3) : 0;
    float coef = logits[(size_t)wid * H + hh] / denom[(size_t)d * H + hh];
    int f0 = lane * 4;
    float4 lv = *(const float4*)(xl + (size_t)s * 128 + f0);
    float* o = acc + (size_t)d * 128 + f0;
    atomicAdd(o + 0, lv.x * coef);
    atomicAdd(o + 1, lv.y * coef);
    atomicAdd(o + 2, lv.z * coef);
    atomicAdd(o + 3, lv.w * coef);
}

// -------------------- pass E: +bias, LayerNorm, ReLU (wave per node) --------
__global__ __launch_bounds__(256)
void bias_ln_relu(const float* __restrict__ acc, const float* __restrict__ cb,
                  const float* __restrict__ gm, const float* __restrict__ bt,
                  float* __restrict__ out, int nN) {
    int wid  = (blockIdx.x * blockDim.x + threadIdx.x) >> 5;
    int lane = threadIdx.x & 31;
    if (wid >= nN) return;
    int f0 = lane * 4;
    float4 v = *(const float4*)(acc + (size_t)wid * 128 + f0);
    v.x += cb[f0 + 0]; v.y += cb[f0 + 1]; v.z += cb[f0 + 2]; v.w += cb[f0 + 3];
    float s = v.x + v.y + v.z + v.w;
    #pragma unroll
    for (int m = 1; m < 32; m <<= 1) s += __shfl_xor(s, m, 32);
    float mu = s * (1.f / 128.f);
    float dx = v.x - mu, dy = v.y - mu, dz = v.z - mu, dw = v.w - mu;
    float q = dx * dx + dy * dy + dz * dz + dw * dw;
    #pragma unroll
    for (int m = 1; m < 32; m <<= 1) q += __shfl_xor(q, m, 32);
    float inv = rsqrtf(q * (1.f / 128.f) + 1e-5f);
    float4 r;
    r.x = fmaxf(dx * inv * gm[f0 + 0] + bt[f0 + 0], 0.f);
    r.y = fmaxf(dy * inv * gm[f0 + 1] + bt[f0 + 1], 0.f);
    r.z = fmaxf(dz * inv * gm[f0 + 2] + bt[f0 + 2], 0.f);
    r.w = fmaxf(dw * inv * gm[f0 + 3] + bt[f0 + 3], 0.f);
    *(float4*)(out + (size_t)wid * 128 + f0) = r;
}

// ---------------------------------------------------------------------------
extern "C" void kernel_launch(void* const* d_in, const int* in_sizes, int n_in,
                              void* d_out, int out_size, void* d_ws, size_t ws_size,
                              hipStream_t stream) {
    const float* x  = (const float*)d_in[0];
    const int*   ei = (const int*)d_in[1];
    const int nN = in_sizes[0] / 128;
    const int nE = in_sizes[1] / 2;
    const int* src = ei;
    const int* dst = ei + nE;

    // parameter pointers, setup_inputs order
    const float* Wl[3] = {(const float*)d_in[2],  (const float*)d_in[10], (const float*)d_in[18]};
    const float* bl[3] = {(const float*)d_in[3],  (const float*)d_in[11], (const float*)d_in[19]};
    const float* Wr[3] = {(const float*)d_in[4],  (const float*)d_in[12], (const float*)d_in[20]};
    const float* br[3] = {(const float*)d_in[5],  (const float*)d_in[13], (const float*)d_in[21]};
    const float* av[3] = {(const float*)d_in[6],  (const float*)d_in[14], (const float*)d_in[22]};
    const float* cv[3] = {(const float*)d_in[7],  (const float*)d_in[15], (const float*)d_in[23]};
    const float* gv[3] = {(const float*)d_in[8],  (const float*)d_in[16], (const float*)d_in[24]};
    const float* bv[3] = {(const float*)d_in[9],  (const float*)d_in[17], (const float*)d_in[25]};

    // workspace layout
    char* ws = (char*)d_ws;
    const size_t NB = (size_t)nN * 128 * sizeof(float);
    const size_t EA = (size_t)nE + (size_t)nN;
    float* xl     = (float*)(ws);
    float* xr     = (float*)(ws + NB);
    float* acc    = (float*)(ws + 2 * NB);
    float* hbuf   = (float*)(ws + 3 * NB);
    float* logits = (float*)(ws + 4 * NB);
    size_t LB = (EA * 4 * sizeof(float) + 255) & ~(size_t)255;
    float* mmax  = (float*)(ws + 4 * NB + LB);
    size_t MB = ((size_t)nN * 4 * sizeof(float) + 255) & ~(size_t)255;
    float* denom = (float*)((char*)mmax + MB);
    unsigned short* wtl = (unsigned short*)((char*)denom + MB);
    unsigned short* wtr = wtl + 128 * 128;

    const int gemmBlocks = (nN + 127) / 128;
    const int edgeBlocks = (int)((EA + 7) / 8);        // 8 waves / 256-thr block
    const int nodeBlocks = (nN + 7) / 8;

    auto run_layer = [&](const float* inF, int L, int H, float* outF) {
        conv_w_bf16t<<<64, 256, 0, stream>>>(Wl[L], wtl);
        conv_w_bf16t<<<64, 256, 0, stream>>>(Wr[L], wtr);
        gemm128_wmma_bf16<<<gemmBlocks, 256, 0, stream>>>(inF, wtl, bl[L], xl, nN);
        gemm128_wmma_bf16<<<gemmBlocks, 256, 0, stream>>>(inF, wtr, br[L], xr, nN);

        hipMemsetAsync(acc, 0, NB, stream);
        hipMemsetAsync(denom, 0, (size_t)nN * H * sizeof(float), stream);
        int fillN = nN * H;
        fill_f32<<<(fillN + 255) / 256, 256, 0, stream>>>(mmax, -__builtin_inff(), fillN);

        if (H == 4) {
            edge_attn_max<4><<<edgeBlocks, 256, 0, stream>>>(xl, xr, src, dst, av[L],
                                                             logits, mmax, nE, nN);
            int ct = (int)(EA * 4);
            edge_softmax_num<4><<<(ct + 255) / 256, 256, 0, stream>>>(dst, logits, mmax,
                                                                     denom, nE, nN);
            edge_scatter<4><<<edgeBlocks, 256, 0, stream>>>(xl, src, dst, logits, denom,
                                                            acc, nE, nN);
        } else {
            edge_attn_max<1><<<edgeBlocks, 256, 0, stream>>>(xl, xr, src, dst, av[L],
                                                             logits, mmax, nE, nN);
            int ct = (int)EA;
            edge_softmax_num<1><<<(ct + 255) / 256, 256, 0, stream>>>(dst, logits, mmax,
                                                                     denom, nE, nN);
            edge_scatter<1><<<edgeBlocks, 256, 0, stream>>>(xl, src, dst, logits, denom,
                                                            acc, nE, nN);
        }
        bias_ln_relu<<<nodeBlocks, 256, 0, stream>>>(acc, cv[L], gv[L], bv[L], outF, nN);
    };

    run_layer(x,    0, 4, hbuf);
    run_layer(hbuf, 1, 4, hbuf);            // safe: GEMMs consume hbuf before overwrite
    run_layer(hbuf, 2, 1, (float*)d_out);
}
